// Attention_35450660061731
// MI455X (gfx1250) — matrix operand order repaired and checked by reading, not compiled, not run on hardware
//
#include <hip/hip_runtime.h>

// ---------------------------------------------------------------------------
// Types for CDNA5 WMMA (wave32, v_wmma_f32_16x16x32_bf16)
// ---------------------------------------------------------------------------
typedef __attribute__((ext_vector_type(16))) __bf16 v16bf;
typedef __attribute__((ext_vector_type(8)))  float  v8f;
typedef __attribute__((ext_vector_type(4)))  float  f32x4;
typedef __attribute__((ext_vector_type(4)))  unsigned int u32x4;
typedef __attribute__((ext_vector_type(2)))  unsigned int u32x2;

union Frag {
    v16bf v;
    u32x4 q[2];
};

__device__ __forceinline__ v8f wmma_bf16(const Frag& a, const Frag& b, v8f c) {
    return __builtin_amdgcn_wmma_f32_16x16x32_bf16(
        /*neg_a=*/false, a.v, /*neg_b=*/false, b.v,
        /*c_mod=*/(short)0, c, /*reuse_a=*/false, /*reuse_b=*/false);
}

__device__ __forceinline__ unsigned short bf16_rn(float f) {
    unsigned int u = __float_as_uint(f);
    u += 0x7FFFu + ((u >> 16) & 1u);     // round-to-nearest-even
    return (unsigned short)(u >> 16);
}

__device__ __forceinline__ u32x2 pack4_bf16(f32x4 v) {
    u32x2 r;
    r.x = (unsigned int)bf16_rn(v.x) | ((unsigned int)bf16_rn(v.y) << 16);
    r.y = (unsigned int)bf16_rn(v.z) | ((unsigned int)bf16_rn(v.w) << 16);
    return r;
}

// ---------------------------------------------------------------------------
// Problem constants
// ---------------------------------------------------------------------------
#define BATCH   16
#define SEQ     577
#define CH      1024
#define HEADS   16
#define HDIM    64
#define NPAD    608                       // 19 key-blocks of 32
#define MROWS   (BATCH * SEQ)             // 9232
#define SCALE   0.125f                    // 1/sqrt(64)

// ===========================================================================
// Kernel 1: QKV GEMM.  128(M) x 64(N) tile, K-step 32, double-buffered LDS,
// register-pipelined float4 global loads (no per-element predication: rows
// are clamped; clamped rows only feed outputs that are never stored).
// Epilogue scatters bf16 into q[B,H,NPAD,64], k[B,H,NPAD,64], vT[B,H,64,NPAD].
// ===========================================================================
__global__ __launch_bounds__(256) void qkv_gemm_kernel(
    const float* __restrict__ x, const float* __restrict__ w_qkv,
    const float* __restrict__ b_qkv,
    unsigned short* __restrict__ qO, unsigned short* __restrict__ kO,
    unsigned short* __restrict__ vT)
{
    __shared__ unsigned short As[2][128 * 32];   // [row][k], double buffered
    __shared__ unsigned short Bs[2][64 * 32];    // [col][k]

    const int tid  = threadIdx.x;
    const int lane = tid & 31;
    const int wave = tid >> 5;
    const int wy   = wave >> 1;               // 0..3  (M direction, 32 rows)
    const int wx   = wave & 1;                // 0..1  (N direction, 32 cols)
    const int m0   = blockIdx.y * 128;
    const int n0   = blockIdx.x * 64;

    const int mrow   = lane & 15;
    const int kbaseA = (lane < 16) ? 0 : 8;
    const int klaneB = (lane < 16) ? 0 : 8;

    // Per-thread staging geometry: A = 4 float4 chunks, B = 2 float4 chunks
    int aRow[4], aKq[4];
    size_t aAddr[4];
    #pragma unroll
    for (int i = 0; i < 4; ++i) {
        int c = i * 256 + tid;                // 0..1023
        aRow[i] = c >> 3;                     // 0..127
        aKq[i]  = (c & 7) * 4;                // 0,4,...,28
        int gr = m0 + aRow[i];
        if (gr > MROWS - 1) gr = MROWS - 1;   // clamp: safe, row never stored
        aAddr[i] = (size_t)gr * CH + aKq[i];
    }
    int bRow[2], bKq[2];
    size_t bAddr[2];
    #pragma unroll
    for (int i = 0; i < 2; ++i) {
        int c = i * 256 + tid;                // 0..511
        bRow[i] = c >> 3;                     // 0..63
        bKq[i]  = (c & 7) * 4;
        bAddr[i] = (size_t)(n0 + bRow[i]) * CH + bKq[i];
    }

    f32x4 aReg[4], bReg[2];
    #pragma unroll
    for (int i = 0; i < 4; ++i) aReg[i] = *(const f32x4*)&x[aAddr[i]];
    #pragma unroll
    for (int i = 0; i < 2; ++i) bReg[i] = *(const f32x4*)&w_qkv[bAddr[i]];

    v8f acc[2][2] = {};
    int p = 0;

    for (int kt = 0; kt < CH / 32; ++kt) {
        // Commit pipelined registers into LDS buffer p (packed bf16, b64)
        #pragma unroll
        for (int i = 0; i < 4; ++i)
            *(u32x2*)&As[p][aRow[i] * 32 + aKq[i]] = pack4_bf16(aReg[i]);
        #pragma unroll
        for (int i = 0; i < 2; ++i)
            *(u32x2*)&Bs[p][bRow[i] * 32 + bKq[i]] = pack4_bf16(bReg[i]);
        __syncthreads();

        // Issue next tile's global loads early (overlap with WMMA below)
        if (kt + 1 < CH / 32) {
            int k0n = (kt + 1) * 32;
            #pragma unroll
            for (int i = 0; i < 4; ++i) aReg[i] = *(const f32x4*)&x[aAddr[i] + k0n];
            #pragma unroll
            for (int i = 0; i < 2; ++i) bReg[i] = *(const f32x4*)&w_qkv[bAddr[i] + k0n];
        }

        Frag a[2], b[2];
        #pragma unroll
        for (int ms = 0; ms < 2; ++ms) {
            const unsigned short* ap = &As[p][(wy * 32 + ms * 16 + mrow) * 32 + kbaseA];
            a[ms].q[0] = *(const u32x4*)(ap);
            a[ms].q[1] = *(const u32x4*)(ap + 16);
        }
        #pragma unroll
        for (int ns = 0; ns < 2; ++ns) {
            const unsigned short* bp = &Bs[p][(wx * 32 + ns * 16 + mrow) * 32 + klaneB];
            b[ns].q[0] = *(const u32x4*)(bp);
            b[ns].q[1] = *(const u32x4*)(bp + 16);
        }
        #pragma unroll
        for (int ms = 0; ms < 2; ++ms)
            #pragma unroll
            for (int ns = 0; ns < 2; ++ns)
                acc[ms][ns] = wmma_bf16(a[ms], b[ns], acc[ms][ns]);
        p ^= 1;
    }

    // Epilogue: add bias, scatter bf16 into q / k / vT layouts
    const int roff = (lane < 16) ? 0 : 8;
    const int col  = lane & 15;
    #pragma unroll
    for (int ms = 0; ms < 2; ++ms) {
        #pragma unroll
        for (int ns = 0; ns < 2; ++ns) {
            int gc = n0 + wx * 32 + ns * 16 + col;
            float bias  = b_qkv[gc];
            int which = gc >> 10;
            int h     = (gc >> 6) & 15;
            int d     = gc & 63;
            #pragma unroll
            for (int r = 0; r < 8; ++r) {
                int gr = m0 + wy * 32 + ms * 16 + r + roff;
                if (gr < MROWS) {
                    int bb = gr / SEQ;
                    int nn = gr - bb * SEQ;
                    unsigned short hv = bf16_rn(acc[ms][ns][r] + bias);
                    if (which == 0)
                        qO[(((size_t)(bb * HEADS + h)) * NPAD + nn) * HDIM + d] = hv;
                    else if (which == 1)
                        kO[(((size_t)(bb * HEADS + h)) * NPAD + nn) * HDIM + d] = hv;
                    else
                        vT[(((size_t)(bb * HEADS + h)) * HDIM + d) * NPAD + nn] = hv;
                }
            }
        }
    }
}

// ===========================================================================
// Kernel 2: flash attention.  One wave = 32 query rows (2 M-tiles): K and V
// fragments are loaded once per 32-key block and reused for both M-tiles.
// S = Q K^T * scale (8 wmma), online softmax, O += P V (8 wmma via vT).
// ===========================================================================
__global__ __launch_bounds__(256) void attn_kernel(
    const unsigned short* __restrict__ q, const unsigned short* __restrict__ k,
    const unsigned short* __restrict__ vT, unsigned short* __restrict__ ctx)
{
    __shared__ unsigned short Pst[8 * 32 * 32];   // per-wave 32x32 P staging

    const int tid  = threadIdx.x;
    const int lane = tid & 31;
    const int wave = tid >> 5;
    const int bh   = blockIdx.x;                  // 0..255
    const int b    = bh >> 4;
    const int h    = bh & 15;
    const int tile = blockIdx.y * 8 + wave;       // 0..23
    const int rowbase = tile * 32;
    if (rowbase >= SEQ) return;                   // no barriers in this kernel

    const int m      = lane & 15;
    const int kbaseA = (lane < 16) ? 0 : 8;
    const int klaneB = (lane < 16) ? 0 : 8;
    unsigned short* pw = &Pst[wave * 1024];

    // Resident Q fragments: [mtile][d-chunk]
    Frag aq[2][2];
    #pragma unroll
    for (int mt = 0; mt < 2; ++mt) {
        const unsigned short* qbase =
            q + ((size_t)bh * NPAD + rowbase + mt * 16 + m) * HDIM;
        #pragma unroll
        for (int dc = 0; dc < 2; ++dc) {
            aq[mt][dc].q[0] = *(const u32x4*)(qbase + dc * 32 + kbaseA);
            aq[mt][dc].q[1] = *(const u32x4*)(qbase + dc * 32 + kbaseA + 16);
        }
    }

    float m_run[2][8], l_run[2][8];
    #pragma unroll
    for (int mt = 0; mt < 2; ++mt)
        #pragma unroll
        for (int r = 0; r < 8; ++r) { m_run[mt][r] = -INFINITY; l_run[mt][r] = 0.0f; }
    v8f oacc[2][4] = {};

    for (int j = 0; j < NPAD / 32; ++j) {
        const int keybase = j * 32;

        // ---- S = Q K^T * scale, masked; K frags shared across both M-tiles
        v8f S[2][2];
        #pragma unroll
        for (int kt = 0; kt < 2; ++kt) {
            int key = keybase + kt * 16 + m;
            const unsigned short* kb = k + ((size_t)bh * NPAD + key) * HDIM;
            Frag bk[2];
            #pragma unroll
            for (int dc = 0; dc < 2; ++dc) {
                bk[dc].q[0] = *(const u32x4*)(kb + dc * 32 + klaneB);
                bk[dc].q[1] = *(const u32x4*)(kb + dc * 32 + klaneB + 16);
            }
            #pragma unroll
            for (int mt = 0; mt < 2; ++mt) {
                v8f sacc = {};
                #pragma unroll
                for (int dc = 0; dc < 2; ++dc)
                    sacc = wmma_bf16(aq[mt][dc], bk[dc], sacc);
                S[mt][kt] = sacc * SCALE;
                if (key >= SEQ) {
                    #pragma unroll
                    for (int e = 0; e < 8; ++e) S[mt][kt][e] = -1e30f;
                }
            }
        }

        // ---- online softmax + stage P (bf16) into per-wave LDS ----
        #pragma unroll
        for (int mt = 0; mt < 2; ++mt) {
            #pragma unroll
            for (int r = 0; r < 8; ++r) {
                float vm = fmaxf(S[mt][0][r], S[mt][1][r]);
                vm = fmaxf(vm, __shfl_xor(vm, 1, 32));
                vm = fmaxf(vm, __shfl_xor(vm, 2, 32));
                vm = fmaxf(vm, __shfl_xor(vm, 4, 32));
                vm = fmaxf(vm, __shfl_xor(vm, 8, 32));
                float mn  = fmaxf(m_run[mt][r], vm);
                float fac = __expf(m_run[mt][r] - mn);
                float p0  = __expf(S[mt][0][r] - mn);
                float p1  = __expf(S[mt][1][r] - mn);
                float rs  = p0 + p1;
                rs += __shfl_xor(rs, 1, 32);
                rs += __shfl_xor(rs, 2, 32);
                rs += __shfl_xor(rs, 4, 32);
                rs += __shfl_xor(rs, 8, 32);
                l_run[mt][r] = l_run[mt][r] * fac + rs;
                m_run[mt][r] = mn;
                #pragma unroll
                for (int t = 0; t < 4; ++t) oacc[mt][t][r] *= fac;
                int rr = mt * 16 + r + ((lane < 16) ? 0 : 8);
                pw[rr * 32 + m]      = bf16_rn(p0);
                pw[rr * 32 + 16 + m] = bf16_rn(p1);
            }
        }

        // ---- O += P V  (V frags shared across both M-tiles) ----
        Frag ap[2];
        #pragma unroll
        for (int mt = 0; mt < 2; ++mt) {
            const unsigned short* pp = pw + (mt * 16 + m) * 32 + kbaseA;
            ap[mt].q[0] = *(const u32x4*)(pp);
            ap[mt].q[1] = *(const u32x4*)(pp + 16);
        }
        #pragma unroll
        for (int dt = 0; dt < 4; ++dt) {
            const unsigned short* vb =
                vT + ((size_t)bh * HDIM + dt * 16 + m) * NPAD + keybase + klaneB;
            Frag bv;
            bv.q[0] = *(const u32x4*)(vb);
            bv.q[1] = *(const u32x4*)(vb + 16);
            #pragma unroll
            for (int mt = 0; mt < 2; ++mt)
                oacc[mt][dt] = wmma_bf16(ap[mt], bv, oacc[mt][dt]);
        }

        // Prefetch next key block (global_prefetch_b8)
        if (j + 1 < NPAD / 32)
            __builtin_prefetch(k + ((size_t)bh * NPAD + keybase + 32 + m) * HDIM, 0, 1);
    }

    // ---- epilogue: O /= l, store bf16 ctx[B*SEQ][CH] ----
    const int roff = (lane < 16) ? 0 : 8;
    #pragma unroll
    for (int mt = 0; mt < 2; ++mt) {
        #pragma unroll
        for (int r = 0; r < 8; ++r) {
            int row = rowbase + mt * 16 + r + roff;
            if (row < SEQ) {
                float inv = 1.0f / l_run[mt][r];
                #pragma unroll
                for (int dt = 0; dt < 4; ++dt) {
                    int colc = h * HDIM + dt * 16 + m;
                    ctx[((size_t)(b * SEQ + row)) * CH + colc] =
                        bf16_rn(oacc[mt][dt][r] * inv);
                }
            }
        }
    }
}

// ===========================================================================
// Kernel 3: projection GEMM (same pipelined/double-buffered scheme;
// A already bf16 -> direct b128 staging, B fp32->bf16).
// ===========================================================================
__global__ __launch_bounds__(256) void proj_gemm_kernel(
    const unsigned short* __restrict__ ctx, const float* __restrict__ w_proj,
    const float* __restrict__ b_proj, float* __restrict__ out)
{
    __shared__ unsigned short As[2][128 * 32];
    __shared__ unsigned short Bs[2][64 * 32];

    const int tid  = threadIdx.x;
    const int lane = tid & 31;
    const int wave = tid >> 5;
    const int wy   = wave >> 1;
    const int wx   = wave & 1;
    const int m0   = blockIdx.y * 128;
    const int n0   = blockIdx.x * 64;

    const int mrow   = lane & 15;
    const int kbaseA = (lane < 16) ? 0 : 8;
    const int klaneB = (lane < 16) ? 0 : 8;

    // A: 2 chunks of 8 bf16 halves per thread; B: 2 chunks of 4 floats
    int aRow[2], aKq[2];
    size_t aAddr[2];
    #pragma unroll
    for (int i = 0; i < 2; ++i) {
        int c = i * 256 + tid;                // 0..511
        aRow[i] = c >> 2;                     // 0..127
        aKq[i]  = (c & 3) * 8;                // 0,8,16,24
        int gr = m0 + aRow[i];
        if (gr > MROWS - 1) gr = MROWS - 1;   // clamp
        aAddr[i] = (size_t)gr * CH + aKq[i];
    }
    int bRow[2], bKq[2];
    size_t bAddr[2];
    #pragma unroll
    for (int i = 0; i < 2; ++i) {
        int c = i * 256 + tid;
        bRow[i] = c >> 3;
        bKq[i]  = (c & 7) * 4;
        bAddr[i] = (size_t)(n0 + bRow[i]) * CH + bKq[i];
    }

    u32x4 aReg[2];
    f32x4 bReg[2];
    #pragma unroll
    for (int i = 0; i < 2; ++i) aReg[i] = *(const u32x4*)&ctx[aAddr[i]];
    #pragma unroll
    for (int i = 0; i < 2; ++i) bReg[i] = *(const f32x4*)&w_proj[bAddr[i]];

    v8f acc[2][2] = {};
    int p = 0;

    for (int kt = 0; kt < CH / 32; ++kt) {
        #pragma unroll
        for (int i = 0; i < 2; ++i)
            *(u32x4*)&As[p][aRow[i] * 32 + aKq[i]] = aReg[i];
        #pragma unroll
        for (int i = 0; i < 2; ++i)
            *(u32x2*)&Bs[p][bRow[i] * 32 + bKq[i]] = pack4_bf16(bReg[i]);
        __syncthreads();

        if (kt + 1 < CH / 32) {
            int k0n = (kt + 1) * 32;
            #pragma unroll
            for (int i = 0; i < 2; ++i) aReg[i] = *(const u32x4*)&ctx[aAddr[i] + k0n];
            #pragma unroll
            for (int i = 0; i < 2; ++i) bReg[i] = *(const f32x4*)&w_proj[bAddr[i] + k0n];
        }

        Frag a[2], b[2];
        #pragma unroll
        for (int ms = 0; ms < 2; ++ms) {
            const unsigned short* ap = &As[p][(wy * 32 + ms * 16 + mrow) * 32 + kbaseA];
            a[ms].q[0] = *(const u32x4*)(ap);
            a[ms].q[1] = *(const u32x4*)(ap + 16);
        }
        #pragma unroll
        for (int ns = 0; ns < 2; ++ns) {
            const unsigned short* bp = &Bs[p][(wx * 32 + ns * 16 + mrow) * 32 + klaneB];
            b[ns].q[0] = *(const u32x4*)(bp);
            b[ns].q[1] = *(const u32x4*)(bp + 16);
        }
        #pragma unroll
        for (int ms = 0; ms < 2; ++ms)
            #pragma unroll
            for (int ns = 0; ns < 2; ++ns)
                acc[ms][ns] = wmma_bf16(a[ms], b[ns], acc[ms][ns]);
        p ^= 1;
    }

    const int roff = (lane < 16) ? 0 : 8;
    const int col  = lane & 15;
    #pragma unroll
    for (int ms = 0; ms < 2; ++ms) {
        #pragma unroll
        for (int ns = 0; ns < 2; ++ns) {
            int gc = n0 + wx * 32 + ns * 16 + col;
            float bias = b_proj[gc];
            #pragma unroll
            for (int r = 0; r < 8; ++r) {
                int gr = m0 + wy * 32 + ms * 16 + r + roff;
                if (gr < MROWS)
                    out[(size_t)gr * CH + gc] = acc[ms][ns][r] + bias;
            }
        }
    }
}

// ===========================================================================
// Host-side launch
// ===========================================================================
extern "C" void kernel_launch(void* const* d_in, const int* in_sizes, int n_in,
                              void* d_out, int out_size, void* d_ws, size_t ws_size,
                              hipStream_t stream) {
    (void)in_sizes; (void)n_in; (void)out_size; (void)ws_size;
    const float* x      = (const float*)d_in[0];
    const float* w_qkv  = (const float*)d_in[1];
    const float* b_qkv  = (const float*)d_in[2];
    const float* w_proj = (const float*)d_in[3];
    const float* b_proj = (const float*)d_in[4];
    float* out = (float*)d_out;

    // Workspace layout (bf16 halves):
    //   q [B,H,NPAD,64]  k [B,H,NPAD,64]  vT [B,H,64,NPAD]  ctx [B*SEQ, CH]
    const size_t QKV_ELEMS = (size_t)BATCH * HEADS * NPAD * HDIM;   // 9,961,472
    unsigned short* qb  = (unsigned short*)d_ws;
    unsigned short* kb  = qb + QKV_ELEMS;
    unsigned short* vt  = kb + QKV_ELEMS;
    unsigned short* ctx = vt + QKV_ELEMS;

    // 1) QKV projection:  grid = (3072/64, ceil(9232/128))
    qkv_gemm_kernel<<<dim3(48, 73), 256, 0, stream>>>(x, w_qkv, b_qkv, qb, kb, vt);
    // 2) attention: grid = (B*H, ceil(19 row-tiles of 32 / 8 waves))
    attn_kernel<<<dim3(BATCH * HEADS, 3), 256, 0, stream>>>(qb, kb, vt, ctx);
    // 3) output projection: grid = (1024/64, ceil(9232/128))
    proj_gemm_kernel<<<dim3(16, 73), 256, 0, stream>>>(ctx, w_proj, b_proj, out);
}
